// AttentionBlock_35485019799765
// MI455X (gfx1250) — matrix operand order repaired and checked by reading, not compiled, not run on hardware
//
#include <hip/hip_runtime.h>
#include <stdint.h>

typedef __bf16 bf16;
typedef __attribute__((ext_vector_type(16))) __bf16 v16bf;
typedef __attribute__((ext_vector_type(8)))  __bf16 v8bf;
typedef __attribute__((ext_vector_type(8)))  float  v8f;

#define SEQ       2048
#define DIM       64
#define BLOCK_M   128     // 8 waves x 16 query rows
#define TILE_N    64      // keys per LDS tile
#define NTILES    (SEQ / TILE_N)
#define NEG_BIG   (-1.0e9f)
#define LDSTR     72      // 64 + 8 bf16 pad -> 144B row stride (16B aligned, conflict-free)

__global__ __launch_bounds__(256) void flash_attn_f32_bf16wmma(
    const float* __restrict__ Q, const float* __restrict__ K,
    const float* __restrict__ V, const uint8_t* __restrict__ mask,
    float* __restrict__ Out)
{
    __shared__ bf16 Kl[2][TILE_N][LDSTR];   // [buf][key][d]
    __shared__ bf16 Vt[2][DIM][LDSTR];      // [buf][d][key]  (transposed at fill)
    __shared__ bf16 Pl[8][16][LDSTR];       // per-wave P staging [m][key_local]

    const int tid  = threadIdx.x;
    const int wave = tid >> 5;
    const int lane = tid & 31;
    const int half = lane >> 4;     // which 16-lane group
    const int n    = lane & 15;     // N column (B/C layouts) == M row (A layout)

    const int bh    = blockIdx.y;
    const int qbase = blockIdx.x * BLOCK_M + wave * 16;

    const float* Qb = Q + (size_t)bh * SEQ * DIM;
    const float* Kb = K + (size_t)bh * SEQ * DIM;
    const float* Vb = V + (size_t)bh * SEQ * DIM;

    // ---- Q tile (16x64) -> WMMA A layout, pre-scaled by 1/sqrt(D) = 0.125
    v16bf qa[2];
    #pragma unroll
    for (int c = 0; c < 2; ++c) {
        const float* qrow = Qb + (size_t)(qbase + n) * DIM + c * 32;
        #pragma unroll
        for (int e = 0; e < 16; ++e) {
            const int k = (half << 4) + e;
            qa[c][e] = (bf16)(qrow[k] * 0.125f);
        }
    }

    v8f o[4] = {};                  // O accumulators (C layout), 16x64 fp32
    float rowM[8], rowL[8];
    #pragma unroll
    for (int v = 0; v < 8; ++v) { rowM[v] = -1.0e30f; rowL[v] = 0.0f; }

    // ---- register staging for the K/V tile fill (double-buffered LDS)
    float4 kreg[4], vreg[4];

    auto load_tile = [&](int kt) {
        const float4* K4 = (const float4*)(Kb + (size_t)kt * TILE_N * DIM);
        const float4* V4 = (const float4*)(Vb + (size_t)kt * TILE_N * DIM);
        #pragma unroll
        for (int i = 0; i < 4; ++i) {
            kreg[i] = K4[tid + i * 256];
            vreg[i] = V4[tid + i * 256];
        }
    };
    auto store_tile = [&](int b) {
        #pragma unroll
        for (int i = 0; i < 4; ++i) {
            const int j   = tid + i * 256;   // 0..1023 float4 index
            const int row = j >> 4;          // key within tile
            const int c4  = (j & 15) << 2;   // d base
            Kl[b][row][c4 + 0] = (bf16)kreg[i].x;
            Kl[b][row][c4 + 1] = (bf16)kreg[i].y;
            Kl[b][row][c4 + 2] = (bf16)kreg[i].z;
            Kl[b][row][c4 + 3] = (bf16)kreg[i].w;
            Vt[b][c4 + 0][row] = (bf16)vreg[i].x;
            Vt[b][c4 + 1][row] = (bf16)vreg[i].y;
            Vt[b][c4 + 2][row] = (bf16)vreg[i].z;
            Vt[b][c4 + 3][row] = (bf16)vreg[i].w;
        }
    };

    // prologue: stage tile 0
    load_tile(0);
    store_tile(0);
    __syncthreads();

    for (int kt = 0; kt < NTILES; ++kt) {
        const int buf = kt & 1;
        const int keybase = kt * TILE_N;

        if (kt + 2 < NTILES) {   // pull tile kt+2 toward L2: global_prefetch_b8
            __builtin_prefetch(Kb + (size_t)(kt + 2) * TILE_N * DIM + tid * 16, 0, 1);
            __builtin_prefetch(Vb + (size_t)(kt + 2) * TILE_N * DIM + tid * 16, 0, 1);
        }

        // ---- S = Q * K^T : per K-chunk, preload all 4 B operands, then 4 WMMAs
        v8f s[4];
        #pragma unroll
        for (int c = 0; c < 2; ++c) {
            v16bf bv[4];
            #pragma unroll
            for (int t = 0; t < 4; ++t) {
                const bf16* bp = &Kl[buf][t * 16 + n][c * 32 + half * 16];
                v8bf blo = *(const v8bf*)bp;
                v8bf bhi = *(const v8bf*)(bp + 8);
                bv[t] = __builtin_shufflevector(blo, bhi,
                    0,1,2,3,4,5,6,7,8,9,10,11,12,13,14,15);
            }
            #pragma unroll
            for (int t = 0; t < 4; ++t) {
                v8f acc = (c == 0) ? (v8f){} : s[t];
                s[t] = __builtin_amdgcn_wmma_f32_16x16x32_bf16(
                         false, qa[c], false, bv[t], (short)0, acc, false, false);
            }
        }

        // ---- issue global loads for next tile (latency covered by softmax+PV)
        if (kt + 1 < NTILES) load_tile(kt + 1);

        // ---- mask (True -> -1e9), C layout: row = v + 8*half, col = t*16 + n
        #pragma unroll
        for (int v = 0; v < 8; ++v) {
            const uint8_t* mq = mask + (size_t)(qbase + v + 8 * half) * SEQ
                                     + keybase + n;
            #pragma unroll
            for (int t = 0; t < 4; ++t)
                if (mq[t * 16]) s[t][v] = NEG_BIG;
        }

        // ---- online softmax (rows distributed: one row per C-layout VGPR slot)
        #pragma unroll
        for (int v = 0; v < 8; ++v) {
            float cm = fmaxf(fmaxf(s[0][v], s[1][v]), fmaxf(s[2][v], s[3][v]));
            #pragma unroll
            for (int off = 8; off > 0; off >>= 1)
                cm = fmaxf(cm, __shfl_xor(cm, off, 32));
            const float nm    = fmaxf(rowM[v], cm);
            const float alpha = __expf(rowM[v] - nm);
            rowM[v] = nm;
            float rs = 0.0f;
            #pragma unroll
            for (int t = 0; t < 4; ++t) {
                const float p = __expf(s[t][v] - nm);
                s[t][v] = p;
                rs += p;
            }
            #pragma unroll
            for (int off = 8; off > 0; off >>= 1)
                rs += __shfl_xor(rs, off, 32);
            rowL[v] = rowL[v] * alpha + rs;
            #pragma unroll
            for (int dt = 0; dt < 4; ++dt) o[dt][v] *= alpha;
        }

        // ---- P (C layout) -> per-wave LDS pad, re-read in A layout.
        // Same-wave LDS ops are serviced in order on CDNA5; hardware DS drain
        // plus compiler memory barrier replaces a full workgroup barrier.
        #pragma unroll
        for (int t = 0; t < 4; ++t)
            #pragma unroll
            for (int v = 0; v < 8; ++v)
                Pl[wave][v + 8 * half][t * 16 + n] = (bf16)s[t][v];

        asm volatile("s_wait_dscnt 0" ::: "memory");

        // ---- O += P * V : per K-chunk, preload A + 4 B operands, then 4 WMMAs
        #pragma unroll
        for (int c = 0; c < 2; ++c) {
            const bf16* pp = &Pl[wave][n][c * 32 + half * 16];
            v8bf plo = *(const v8bf*)pp;          // K = c*32 + half*16 + 0..7
            v8bf phi = *(const v8bf*)(pp + 8);    // K = c*32 + half*16 + 8..15
            v16bf av = __builtin_shufflevector(plo, phi,
                0,1,2,3,4,5,6,7,8,9,10,11,12,13,14,15);
            v16bf bv[4];
            #pragma unroll
            for (int dt = 0; dt < 4; ++dt) {
                const bf16* vp = &Vt[buf][dt * 16 + n][c * 32 + half * 16];
                v8bf vlo = *(const v8bf*)vp;
                v8bf vhi = *(const v8bf*)(vp + 8);
                bv[dt] = __builtin_shufflevector(vlo, vhi,
                    0,1,2,3,4,5,6,7,8,9,10,11,12,13,14,15);
            }
            #pragma unroll
            for (int dt = 0; dt < 4; ++dt)
                o[dt] = __builtin_amdgcn_wmma_f32_16x16x32_bf16(
                          false, av, false, bv[dt], (short)0, o[dt], false, false);
        }

        // ---- commit next tile to the other LDS buffer, publish with one barrier
        if (kt + 1 < NTILES) store_tile(buf ^ 1);
        __syncthreads();
    }

    // ---- normalize and store (coalesced across the 16-lane groups)
    #pragma unroll
    for (int v = 0; v < 8; ++v) {
        const int q = qbase + v + 8 * half;
        const float inv = 1.0f / rowL[v];
        float* orow = Out + ((size_t)bh * SEQ + q) * DIM;
        #pragma unroll
        for (int dt = 0; dt < 4; ++dt)
            orow[dt * 16 + n] = o[dt][v] * inv;
    }
}

extern "C" void kernel_launch(void* const* d_in, const int* in_sizes, int n_in,
                              void* d_out, int out_size, void* d_ws, size_t ws_size,
                              hipStream_t stream) {
    (void)n_in; (void)out_size; (void)d_ws; (void)ws_size;
    const float*   Q = (const float*)d_in[0];
    const float*   K = (const float*)d_in[1];
    const float*   V = (const float*)d_in[2];
    const uint8_t* M = (const uint8_t*)d_in[3];
    float* O = (float*)d_out;

    const int BH = in_sizes[0] / (SEQ * DIM);   // B*H = 64
    dim3 grid(SEQ / BLOCK_M, BH);
    dim3 block(256);
    flash_attn_f32_bf16wmma<<<grid, block, 0, stream>>>(Q, K, V, M, O);
}
